// RNN_33294586478690
// MI455X (gfx1250) — matrix operand order repaired and checked by reading, not compiled, not run on hardware
//
#include <hip/hip_runtime.h>
#include <hip/hip_bf16.h>
#include <stdint.h>

// Problem constants (match reference)
constexpr int V = 32000, E = 512, H = 1024, O = 6;
constexpr int B = 256, T = 128;
constexpr int KDIM = E + H;          // 1536
constexpr int KT = KDIM / 32;        // 48 K-tiles of 32
constexpr int NT = H / 16;           // 64 N-tiles of 16
constexpr int PANEL_NT = 4;          // n-tiles per block (64 columns)
constexpr size_t FRAG = 32 * 16;     // bf16 elems per packed B fragment (1 KB)
constexpr size_t PANEL_ELEMS = (size_t)PANEL_NT * KT * FRAG;   // 98304 bf16
constexpr size_t PANEL_BYTES = PANEL_ELEMS * 2;                // 196608 B LDS

typedef __bf16 bf16;
typedef __attribute__((ext_vector_type(16))) __bf16 v16bf;
typedef __attribute__((ext_vector_type(8)))  __bf16 v8bf;
typedef __attribute__((ext_vector_type(8)))  float  v8f;
typedef __attribute__((ext_vector_type(4)))  uint32_t u32x4;
typedef __attribute__((ext_vector_type(8)))  int      i32x8;
typedef __attribute__((ext_vector_type(4)))  int      i32x4;

// ---------------------------------------------------------------------------
// Prep 1: gather all embeddings: X[t][b][e] = We[e][tok] + be[e]  (bf16)
// ---------------------------------------------------------------------------
__global__ __launch_bounds__(128) void embed_kernel(
    const int* __restrict__ tokens, const float* __restrict__ We,
    const float* __restrict__ be, bf16* __restrict__ X) {
  const int bt = blockIdx.x;          // b*T + t
  const int b  = bt / T;
  const int t  = bt % T;
  const int tok = tokens[bt];
  const size_t dst = ((size_t)t * B + b) * E;
  for (int e = threadIdx.x; e < E; e += blockDim.x) {
    float v = We[(size_t)e * V + tok] + be[e];
    X[dst + e] = (bf16)v;
  }
}

// ---------------------------------------------------------------------------
// Prep 2: pack Wh [H][KDIM] fp32 -> bf16 B-fragments for V_WMMA_F32_16X16X32_BF16.
// Fragment (nt, kt), lane l: 16 contiguous bf16 = Wh[nt*16+(l&15)][kt*32+(l>>4)*16+j].
// ---------------------------------------------------------------------------
__global__ __launch_bounds__(256) void pack_wh_kernel(
    const float* __restrict__ Wh, bf16* __restrict__ WhP) {
  const int tid = blockIdx.x * blockDim.x + threadIdx.x;
  if (tid >= NT * KT * 32) return;
  const int lane = tid & 31;
  const int frag = tid >> 5;          // nt*KT + kt
  const int kt = frag % KT;
  const int nt = frag / KT;
  const int n = nt * 16 + (lane & 15);
  const int kbase = kt * 32 + (lane >> 4) * 16;
  const size_t src = (size_t)n * KDIM + kbase;
  const size_t dst = (size_t)frag * FRAG + (size_t)lane * 16;
#pragma unroll
  for (int j = 0; j < 16; ++j) WhP[dst + j] = (bf16)Wh[src + j];
}

__global__ __launch_bounds__(256) void zero_kernel(uint32_t* __restrict__ p, int n) {
  int i = blockIdx.x * blockDim.x + threadIdx.x;
  if (i < n) p[i] = 0u;
}

// ---------------------------------------------------------------------------
// TDM: stage this block's contiguous 192 KB B panel (WhP + nt0*KT*FRAG) into LDS.
// 1-row tile of 24576 8-byte elements; D# built per CDNA5 ISA 08_async_tensor.
// ---------------------------------------------------------------------------
__device__ __forceinline__ void tdm_load_panel(const bf16* gsrc) {
  const uint64_t ga = (uint64_t)(uintptr_t)gsrc;
  constexpr uint32_t NELEM8 = (uint32_t)(PANEL_BYTES / 8);   // 24576 x 8B

  u32x4 g0;
  g0[0] = 1u;                                   // count=1, is_restore=0
  g0[1] = 0u;                                   // lds_addr = 0 (dynamic LDS base)
  g0[2] = (uint32_t)(ga & 0xFFFFFFFFu);         // global_addr[31:0]
  g0[3] = (uint32_t)((ga >> 32) & 0x01FFFFFFu)  // global_addr[56:32]
        | (2u << 30);                           // type = 2 ("image")

  i32x8 g1;
  g1[0] = (int)(3u << 16);                      // wg_mask=0, data_size=3 (8B)
  g1[1] = (int)((NELEM8 & 0xFFFFu) << 16);      // tensor_dim0[15:0]
  g1[2] = (int)((NELEM8 >> 16) | (1u << 16));   // tensor_dim0[31:16], tensor_dim1=1
  g1[3] = (int)((NELEM8 & 0xFFFFu) << 16);      // tile_dim0 = NELEM8
  g1[4] = (int)1;                               // tile_dim1 = 1, tile_dim2 = 0
  g1[5] = (int)NELEM8;                          // tensor_dim0_stride[31:0]
  g1[6] = 0;                                    // stride[47:32], dim1_stride lo
  g1[7] = 0;
  i32x4 gz4 = {0, 0, 0, 0};
  i32x8 gz8 = {0, 0, 0, 0, 0, 0, 0, 0};

  __builtin_amdgcn_tensor_load_to_lds(g0, g1, gz4, gz4, gz8, 0);
  __builtin_amdgcn_s_wait_tensorcnt(0);
}

// ---------------------------------------------------------------------------
// Recurrence step: Hout = [Xt | Hin] @ Wh^T + bh (bf16 in, fp32 WMMA acc).
// Grid (16,4) x 128 threads; block = 64x64 output, wave = 16x64 tile.
// B panel comes from LDS (ds_load_b128), A rows stream from global.
// ---------------------------------------------------------------------------
__global__ __launch_bounds__(128) void step_kernel(
    const bf16* __restrict__ Xt,   // [B][E]
    const bf16* __restrict__ Hin,  // [B][H]
    const bf16* __restrict__ WhP,  // packed B-fragments [NT*KT*FRAG]
    const float* __restrict__ bh,
    bf16* __restrict__ Hout) {     // [B][H]
  extern __shared__ __align__(16) char smem[];
  const bf16* Blds = (const bf16*)smem;

  const int lane = threadIdx.x & 31;
  const int wave = threadIdx.x >> 5;
  const int half = lane >> 4;
  const int colL = lane & 15;
  const int m0 = blockIdx.y * 64 + wave * 16;
  const int n0 = blockIdx.x * 64;
  const int nt0 = blockIdx.x * PANEL_NT;
  const int rowA = m0 + colL;

  // One wave per block drives the TDM; TENSORcnt is per-wave.
  if (wave == 0) tdm_load_panel(WhP + (size_t)nt0 * KT * FRAG);
  __syncthreads();

  v8f acc0 = {}, acc1 = {}, acc2 = {}, acc3 = {};

  // ---- K = 0..511 : embedding part (A from Xt, ld = E) ----
  for (int kt = 0; kt < E / 32; ++kt) {
    const bf16* pa = Xt + (size_t)rowA * E + kt * 32 + half * 16;
    __builtin_prefetch(pa + 32, 0, 1);
    v8bf lo = *(const v8bf*)pa;
    v8bf hi = *(const v8bf*)(pa + 8);
    v16bf a = __builtin_shufflevector(lo, hi, 0,1,2,3,4,5,6,7,8,9,10,11,12,13,14,15);
    const bf16* pb = Blds + (size_t)kt * FRAG + (size_t)lane * 16;
    v16bf b0 = *(const v16bf*)pb;
    v16bf b1 = *(const v16bf*)(pb + KT * FRAG);
    v16bf b2 = *(const v16bf*)(pb + 2 * KT * FRAG);
    v16bf b3 = *(const v16bf*)(pb + 3 * KT * FRAG);
    acc0 = __builtin_amdgcn_wmma_f32_16x16x32_bf16(false, a, false, b0, (short)0, acc0, false, false);
    acc1 = __builtin_amdgcn_wmma_f32_16x16x32_bf16(false, a, false, b1, (short)0, acc1, false, false);
    acc2 = __builtin_amdgcn_wmma_f32_16x16x32_bf16(false, a, false, b2, (short)0, acc2, false, false);
    acc3 = __builtin_amdgcn_wmma_f32_16x16x32_bf16(false, a, false, b3, (short)0, acc3, false, false);
  }

  // ---- K = 512..1535 : hidden part (A from Hin, ld = H) ----
  for (int kt = E / 32; kt < KT; ++kt) {
    const int k0 = (kt - E / 32) * 32;
    const bf16* pa = Hin + (size_t)rowA * H + k0 + half * 16;
    __builtin_prefetch(pa + 32, 0, 1);
    v8bf lo = *(const v8bf*)pa;
    v8bf hi = *(const v8bf*)(pa + 8);
    v16bf a = __builtin_shufflevector(lo, hi, 0,1,2,3,4,5,6,7,8,9,10,11,12,13,14,15);
    const bf16* pb = Blds + (size_t)kt * FRAG + (size_t)lane * 16;
    v16bf b0 = *(const v16bf*)pb;
    v16bf b1 = *(const v16bf*)(pb + KT * FRAG);
    v16bf b2 = *(const v16bf*)(pb + 2 * KT * FRAG);
    v16bf b3 = *(const v16bf*)(pb + 3 * KT * FRAG);
    acc0 = __builtin_amdgcn_wmma_f32_16x16x32_bf16(false, a, false, b0, (short)0, acc0, false, false);
    acc1 = __builtin_amdgcn_wmma_f32_16x16x32_bf16(false, a, false, b1, (short)0, acc1, false, false);
    acc2 = __builtin_amdgcn_wmma_f32_16x16x32_bf16(false, a, false, b2, (short)0, acc2, false, false);
    acc3 = __builtin_amdgcn_wmma_f32_16x16x32_bf16(false, a, false, b3, (short)0, acc3, false, false);
  }

  // Epilogue: C/D layout — VGPR r holds M = r + half*8, N = n0 + tile*16 + colL.
  const float b0v = bh[n0 + 0  + colL];
  const float b1v = bh[n0 + 16 + colL];
  const float b2v = bh[n0 + 32 + colL];
  const float b3v = bh[n0 + 48 + colL];
#pragma unroll
  for (int r = 0; r < 8; ++r) {
    const int m = m0 + half * 8 + r;
    bf16* prow = Hout + (size_t)m * H + n0;
    prow[colL]      = (bf16)(acc0[r] + b0v);
    prow[16 + colL] = (bf16)(acc1[r] + b1v);
    prow[32 + colL] = (bf16)(acc2[r] + b2v);
    prow[48 + colL] = (bf16)(acc3[r] + b3v);
  }
}

// ---------------------------------------------------------------------------
// Output head: out[b][o] = [Xlast_b | Hfin_b] . Wo[o] + bo[o]   (O = 6)
// ---------------------------------------------------------------------------
__global__ __launch_bounds__(256) void out_kernel(
    const bf16* __restrict__ Xlast, const bf16* __restrict__ Hfin,
    const float* __restrict__ Wo, const float* __restrict__ bo,
    float* __restrict__ out) {
  const int b = blockIdx.x;
  __shared__ float red[O][256];
  float acc[O];
#pragma unroll
  for (int o = 0; o < O; ++o) acc[o] = 0.f;
  for (int k = threadIdx.x; k < KDIM; k += 256) {
    float x = (k < E) ? (float)Xlast[(size_t)b * E + k]
                      : (float)Hfin[(size_t)b * H + (k - E)];
#pragma unroll
    for (int o = 0; o < O; ++o) acc[o] += x * Wo[(size_t)o * KDIM + k];
  }
#pragma unroll
  for (int o = 0; o < O; ++o) red[o][threadIdx.x] = acc[o];
  __syncthreads();
  for (int s = 128; s > 0; s >>= 1) {
    if (threadIdx.x < (unsigned)s)
#pragma unroll
      for (int o = 0; o < O; ++o) red[o][threadIdx.x] += red[o][threadIdx.x + s];
    __syncthreads();
  }
  if (threadIdx.x < O) out[b * O + threadIdx.x] = red[threadIdx.x][0] + bo[threadIdx.x];
}

// ---------------------------------------------------------------------------
extern "C" void kernel_launch(void* const* d_in, const int* in_sizes, int n_in,
                              void* d_out, int out_size, void* d_ws, size_t ws_size,
                              hipStream_t stream) {
  const int*   tokens = (const int*)  d_in[0];
  const float* We     = (const float*)d_in[1];
  const float* be     = (const float*)d_in[2];
  const float* Wh     = (const float*)d_in[3];
  const float* bh     = (const float*)d_in[4];
  const float* Wo     = (const float*)d_in[5];
  const float* bo     = (const float*)d_in[6];
  float* out = (float*)d_out;

  // Workspace: X (33.5 MB) | WhP (3 MB) | Hbuf0/Hbuf1 (0.5 MB each)
  char* ws = (char*)d_ws;
  bf16* X     = (bf16*)ws;
  bf16* WhP   = (bf16*)(ws + (size_t)T * B * E * sizeof(bf16));
  bf16* Hbuf0 = (bf16*)(ws + (size_t)T * B * E * sizeof(bf16)
                           + (size_t)H * KDIM * sizeof(bf16));
  bf16* Hbuf1 = Hbuf0 + (size_t)B * H;

  embed_kernel<<<B * T, 128, 0, stream>>>(tokens, We, be, X);
  pack_wh_kernel<<<(NT * KT * 32 + 255) / 256, 256, 0, stream>>>(Wh, WhP);
  zero_kernel<<<(B * H / 2 + 255) / 256, 256, 0, stream>>>((uint32_t*)Hbuf0, B * H / 2);

  bf16* hin = Hbuf0;
  bf16* hout = Hbuf1;
  for (int t = 0; t < T - 1; ++t) {
    step_kernel<<<dim3(H / 64, B / 64), 128, PANEL_BYTES, stream>>>(
        X + (size_t)t * B * E, hin, WhP, bh, hout);
    bf16* tmp = hin; hin = hout; hout = tmp;
  }
  out_kernel<<<B, 256, 0, stream>>>(X + (size_t)(T - 1) * B * E, hin, Wo, bo, out);
}
